// MSFCN6_CCA_71073118814765
// MI455X (gfx1250) — matrix operand order repaired and checked by reading, not compiled, not run on hardware
//
#include <hip/hip_runtime.h>
#include <cstdint>

// ---------------------------------------------------------------------------
// MI455X (gfx1250) implementation: conv stack as implicit-GEMM over
// V_WMMA_F32_16X16X32_BF16 (wave32), 4 M-tiles per wave so each im2col
// B-fragment feeds 4 WMMAs. CCA attention folded into per-channel mean
// accumulation. All intermediates bf16 channels-last, resident in L2.
// ---------------------------------------------------------------------------

typedef __bf16 bf16t;
typedef __attribute__((ext_vector_type(16))) __bf16 v16bf;
typedef __attribute__((ext_vector_type(8)))  __bf16 v8bf;
typedef __attribute__((ext_vector_type(8)))  float  v8f;

#define DEVFN __device__ __forceinline__

// ------------------------- window gather (c_in) ----------------------------
// c_in[b,h,n] = xp[b, j*32768 + e], t = h*129+n, j = t>>16, e = t&0xFFFF
DEVFN float cin_sample(const float* __restrict__ x, int b, int64_t h, int64_t n) {
  int64_t t = h * 129 + n;
  int64_t j = t >> 16;
  int64_t e = t & 0xFFFF;
  int64_t src = j * 32768 + e;
  return (src < 4194304) ? x[(int64_t)b * 4194304 + src] : 0.0f;
}

__global__ void k_build_cin0(const float* __restrict__ x, bf16t* __restrict__ out) {
  int64_t i = (int64_t)blockIdx.x * blockDim.x + threadIdx.x;
  const int64_t total = (int64_t)2 * 65536 * 129;
  if (i >= total) return;
  int n = (int)(i % 129);
  int64_t r = i / 129;
  int h = (int)(r % 65536);
  int b = (int)(r / 65536);
  out[i] = (bf16t)cin_sample(x, b, h, n);
}

// AvgPool2d(2,2,pad=1,count_include_pad) applied to c_in -> (B,32769,65)
__global__ void k_build_cin1(const float* __restrict__ x, bf16t* __restrict__ out) {
  int64_t i = (int64_t)blockIdx.x * blockDim.x + threadIdx.x;
  const int64_t total = (int64_t)2 * 32769 * 65;
  if (i >= total) return;
  int ow = (int)(i % 65);
  int64_t r = i / 65;
  int oh = (int)(r % 32769);
  int b  = (int)(r / 32769);
  float s = 0.0f;
  for (int dh = 0; dh < 2; ++dh)
    for (int dw = 0; dw < 2; ++dw) {
      int ih = 2 * oh - 1 + dh, iw = 2 * ow - 1 + dw;
      if (ih >= 0 && ih < 65536 && iw >= 0 && iw < 129)
        s += cin_sample(x, b, ih, iw);
    }
  out[i] = (bf16t)(s * 0.25f);
}

// ----------------------- weight repack (f32 -> bf16) -----------------------
// (COUT,CIN,KH,KW) f32 -> row-major (COUT, Kpad) bf16, k = (kh*KW+kw)*CIN + c
__global__ void k_prep_w(const float* __restrict__ w, bf16t* __restrict__ wbf,
                         int COUT, int CIN, int KH, int KW, int Kpad) {
  int i = blockIdx.x * blockDim.x + threadIdx.x;
  if (i >= COUT * Kpad) return;
  int k = i % Kpad, m = i / Kpad;
  int K = CIN * KH * KW;
  float v = 0.0f;
  if (k < K) {
    int c = k % CIN;
    int t = k / CIN;
    int kw = t % KW, kh = t / KW;
    v = w[(((int64_t)m * CIN + c) * KH + kh) * KW + kw];
  }
  wbf[i] = (bf16t)v;
}

// ------------------- implicit-GEMM conv via WMMA bf16 ----------------------
// One wave computes a (16*MT Cout) x 16(pixels) tile; K = CIN*KH*KW (pad 32).
// The im2col B fragment is gathered once per K-step and feeds MT WMMAs.
// Fragment packing: A and B both contiguous, k = kb + 16*half + j (16-bit A 16x32, B 32x16, f32 C/D).
template<int CIN, int COUT, int KH, int KW, int S, int P, bool RELU, int MT>
__global__ __launch_bounds__(128) void k_conv_wmma(
    const bf16t* __restrict__ in, int Hin, int Win,
    const bf16t* __restrict__ wbf, const float* __restrict__ bias,
    bf16t* __restrict__ out, int Hout, int Wout) {
  constexpr int K    = CIN * KH * KW;
  constexpr int Kpad = (K + 31) & ~31;

  const int lane = threadIdx.x & 31;
  const int wave = threadIdx.x >> 5;
  const int half = lane >> 4;      // lane group 0..15 vs 16..31
  const int l16  = lane & 15;
  const int b    = blockIdx.z;
  const int M0   = blockIdx.y * (16 * MT);
  const int Npix = Hout * Wout;
  const int N0   = (blockIdx.x * 4 + wave) * 16;
  if (N0 >= Npix) return;          // wave-uniform; EXEC all-1s at WMMA

  // this lane's output-pixel column (B fragment / store column)
  const int  n       = N0 + l16;
  const bool ncol_ok = (n < Npix);
  const int  oh  = ncol_ok ? (n / Wout) : 0;
  const int  ow  = ncol_ok ? (n % Wout) : 0;
  const int  ih0 = oh * S - P;
  const int  iw0 = ow * S - P;

  const int64_t inB = (int64_t)b * Hin * Win * CIN;
  const bf16t* wrow[MT];
#pragma unroll
  for (int mt = 0; mt < MT; ++mt)
    wrow[mt] = wbf + (int64_t)(M0 + mt * 16 + l16) * Kpad;

  v8f acc[MT];
#pragma unroll
  for (int mt = 0; mt < MT; ++mt) acc[mt] = (v8f){};

  for (int kb = 0; kb < Kpad; kb += 32) {
    // ---- B (im2col): element j -> K row kb + 16*half + j, column n
    v16bf bfv = {};
    if constexpr (CIN % 16 == 0) {
      // one contiguous 16-channel run, 32B-aligned -> vector load
      const int k = kb + 16 * half;
      if (k < K) {
        const int c0 = k % CIN;
        const int t  = k / CIN;
        const int kw = t % KW, kh = t / KW;
        const int ih = ih0 + kh, iw = iw0 + kw;
        if (ncol_ok && ih >= 0 && ih < Hin && iw >= 0 && iw < Win)
          bfv = *(const v16bf*)(in + inB + ((int64_t)ih * Win + iw) * CIN + c0);
      }
    } else {
#pragma unroll
      for (int j = 0; j < 16; ++j) {
        const int k = kb + 16 * half + j;
        bf16t val = (bf16t)0.0f;
        if (ncol_ok && k < K) {
          const int c = k % CIN;
          const int t = k / CIN;
          const int kw = t % KW, kh = t / KW;
          const int ih = ih0 + kh, iw = iw0 + kw;
          if (ih >= 0 && ih < Hin && iw >= 0 && iw < Win)
            val = in[inB + ((int64_t)ih * Win + iw) * CIN + c];
        }
        bfv[j] = val;
      }
    }
    // ---- A (weights): lane row = M0 + mt*16 + l16.
    // half==0: elems 0..15 -> K=kb+j
    // half==1: elems 0..15 -> K=kb+16+j
#pragma unroll
    for (int mt = 0; mt < MT; ++mt) {
      v8bf a0 = *(const v8bf*)(wrow[mt] + kb + 16 * half);
      v8bf a1 = *(const v8bf*)(wrow[mt] + kb + 16 * half + 8);
      v16bf a = __builtin_shufflevector(a0, a1, 0, 1, 2, 3, 4, 5, 6, 7,
                                        8, 9, 10, 11, 12, 13, 14, 15);
      acc[mt] = __builtin_amdgcn_wmma_f32_16x16x32_bf16(
          /*neg_a=*/false, a, /*neg_b=*/false, bfv,
          /*c_mod=*/(short)0, acc[mt], /*reuse_a=*/false, /*reuse_b=*/false);
    }
  }

  // D layout: VGPR r -> row (M-tile base) + r + 8*half, column n
  if (!ncol_ok) return;
  const int64_t outP = (int64_t)b * Npix * COUT + ((int64_t)oh * Wout + ow) * COUT;
#pragma unroll
  for (int mt = 0; mt < MT; ++mt) {
#pragma unroll
    for (int r = 0; r < 8; ++r) {
      const int m = M0 + mt * 16 + 8 * half + r;
      float v = acc[mt][r] + bias[m];
      if (RELU) v = v > 0.0f ? v : 0.0f;
      out[outP + m] = (bf16t)v;
    }
  }
}

// --------------------------- CCA + channel-mean ----------------------------
// Per pixel (b,h,w): softmax over [eH(1..H with -inf diag) | eW(1..W)], then
// accumulate gamma*(outH+outW)[c] + x[c] into featacc[b, chanOff+c].
template<int C, int CQ>
__global__ __launch_bounds__(256) void k_cca(
    const bf16t* __restrict__ xm, const bf16t* __restrict__ q,
    const bf16t* __restrict__ kk, const bf16t* __restrict__ vv,
    const float* __restrict__ gamma, float* __restrict__ featacc,
    int H, int W, int chanOff) {
  extern __shared__ float sm[];
  float* qs  = sm;                 // CQ
  float* att = sm + CQ;            // H + W
  float* red = att + (H + W);      // 256

  const int p = blockIdx.x;        // pixel
  const int b = blockIdx.y;
  const int h = p / W, w = p % W;
  const int tid = threadIdx.x;
  const int HW2 = H + W;

  if (tid < CQ) qs[tid] = (float)q[(((int64_t)b * H + h) * W + w) * CQ + tid];
  __syncthreads();

  // energies
  for (int i = tid; i < HW2; i += 256) {
    float e;
    if (i < H) {
      if (i == h) { e = -1e30f; }
      else {
        const bf16t* kp = kk + (((int64_t)b * H + i) * W + w) * CQ;
        float s = 0.0f;
#pragma unroll
        for (int c = 0; c < CQ; ++c) s += qs[c] * (float)kp[c];
        e = s;
      }
    } else {
      const bf16t* kp = kk + (((int64_t)b * H + h) * W + (i - H)) * CQ;
      float s = 0.0f;
#pragma unroll
      for (int c = 0; c < CQ; ++c) s += qs[c] * (float)kp[c];
      e = s;
    }
    att[i] = e;
  }
  __syncthreads();

  // softmax: max
  float lm = -1e30f;
  for (int i = tid; i < HW2; i += 256) lm = fmaxf(lm, att[i]);
  red[tid] = lm; __syncthreads();
  for (int s = 128; s > 0; s >>= 1) {
    if (tid < s) red[tid] = fmaxf(red[tid], red[tid + s]);
    __syncthreads();
  }
  float mx = red[0]; __syncthreads();
  // exp + sum
  float ls = 0.0f;
  for (int i = tid; i < HW2; i += 256) {
    float e = __expf(att[i] - mx);
    att[i] = e; ls += e;
  }
  red[tid] = ls; __syncthreads();
  for (int s = 128; s > 0; s >>= 1) {
    if (tid < s) red[tid] += red[tid + s];
    __syncthreads();
  }
  const float inv = 1.0f / red[0];
  __syncthreads();

  // aggregation (per channel), fold into mean accumulator
  const float gm = gamma[0] * inv;
  for (int c = tid; c < C; c += 256) {
    float accH = 0.0f;
    const bf16t* vcol = vv + ((int64_t)b * H * W + w) * C + c;   // stride W*C over g
    for (int g = 0; g < H; ++g) accH += att[g] * (float)vcol[(int64_t)g * W * C];
    float accW = 0.0f;
    const bf16t* vrow = vv + (((int64_t)b * H + h) * W) * C + c; // stride C over v
    for (int x = 0; x < W; ++x) accW += att[H + x] * (float)vrow[(int64_t)x * C];
    float res = gm * (accH + accW) +
                (float)xm[(((int64_t)b * H + h) * W + w) * C + c];
    atomicAdd(&featacc[b * 256 + chanOff + c], res);
  }
}

__global__ void k_zero(float* __restrict__ p, int n) {
  int i = blockIdx.x * blockDim.x + threadIdx.x;
  if (i < n) p[i] = 0.0f;
}

// feat = [mean a0 | mean a1]; out = feat @ cls_w.T + cls_b
__global__ void k_classify(const float* __restrict__ featacc,
                           const float* __restrict__ cw, const float* __restrict__ cb,
                           float* __restrict__ out, int n0, int n1) {
  int t = threadIdx.x;
  if (t >= 4) return;
  int b = t >> 1, o = t & 1;
  float s = cb[o];
  for (int j = 0; j < 256; ++j) {
    float f = featacc[b * 256 + j] * (j < 128 ? 1.0f / (float)n0 : 1.0f / (float)n1);
    s += f * cw[o * 256 + j];
  }
  out[b * 2 + o] = s;
}

// ---------------------------------------------------------------------------
extern "C" void kernel_launch(void* const* d_in, const int* in_sizes, int n_in,
                              void* d_out, int out_size, void* d_ws, size_t ws_size,
                              hipStream_t stream) {
  (void)in_sizes; (void)n_in; (void)out_size; (void)ws_size;
  const float* x  = (const float*)d_in[0];
  const float* f0w[4] = {(const float*)d_in[1], (const float*)d_in[3],
                         (const float*)d_in[5], (const float*)d_in[7]};
  const float* f0b[4] = {(const float*)d_in[2], (const float*)d_in[4],
                         (const float*)d_in[6], (const float*)d_in[8]};
  const float* f1w[4] = {(const float*)d_in[9],  (const float*)d_in[11],
                         (const float*)d_in[13], (const float*)d_in[15]};
  const float* f1b[4] = {(const float*)d_in[10], (const float*)d_in[12],
                         (const float*)d_in[14], (const float*)d_in[16]};
  const float* qw = (const float*)d_in[17]; const float* qb = (const float*)d_in[18];
  const float* kw = (const float*)d_in[19]; const float* kb = (const float*)d_in[20];
  const float* vw = (const float*)d_in[21]; const float* vb = (const float*)d_in[22];
  const float* gm = (const float*)d_in[23];
  const float* cw = (const float*)d_in[24]; const float* cb = (const float*)d_in[25];

  char* ws = (char*)d_ws;
  constexpr size_t MB = 1ull << 20;
  // Arena with lifetime-based reuse (peak ~117 MB, fits in L2-resident set).
  bf16t* CIN0 = (bf16t*)(ws + 0);         // (2,65536,129)   33.8MB, dead after f0.c1
  bf16t* L2_0 = (bf16t*)(ws + 0);         // (2,2731,6,128)   8.4MB (reuse)
  bf16t* L3_0 = (bf16t*)(ws + 9  * MB);   // (2,1366,3,256)   4.2MB
  bf16t* C0   = (bf16t*)(ws + 14 * MB);   // (2,1366,3,128)   2.1MB
  bf16t* L1_1 = (bf16t*)(ws + 17 * MB);   // (2,5462,11,64)  15.4MB
  bf16t* CIN1 = (bf16t*)(ws + 34 * MB);   // (2,32769,65)     8.6MB, dead after f1.c1
  bf16t* L2_1 = (bf16t*)(ws + 34 * MB);   // (2,1366,3,128)   2.1MB (reuse)
  bf16t* L3_1 = (bf16t*)(ws + 37 * MB);   // (2,683,2,256)    1.4MB
  bf16t* C1   = (bf16t*)(ws + 39 * MB);   // (2,683,2,128)    0.7MB
  bf16t* Q0   = (bf16t*)(ws + 40 * MB);
  bf16t* K0   = (bf16t*)(ws + 40 * MB + 512 * 1024);
  bf16t* V0   = (bf16t*)(ws + 41 * MB);   // 2.1MB
  bf16t* Q1   = (bf16t*)(ws + 44 * MB);
  bf16t* K1   = (bf16t*)(ws + 44 * MB + 256 * 1024);
  bf16t* V1   = (bf16t*)(ws + 44 * MB + 512 * 1024);
  float* FEAT = (float*)(ws + 46 * MB);   // (2,256) f32
  bf16t* L1_0 = (bf16t*)(ws + 48 * MB);   // (2,10923,22,64) 61.6MB
  bf16t* WBF  = (bf16t*)(ws + 110 * MB);  // packed bf16 weights, 6.7MB

  // weight sub-offsets (bf16 elements)
  bf16t* Wf0_1 = WBF;            bf16t* Wf0_2 = WBF + 10240;
  bf16t* Wf0_3 = WBF + 534528;   bf16t* Wf0_4 = WBF + 1353728;
  bf16t* Wf1_1 = WBF + 1648640;  bf16t* Wf1_2 = WBF + 1658880;
  bf16t* Wf1_3 = WBF + 2183168;  bf16t* Wf1_4 = WBF + 3002368;
  bf16t* Wq    = WBF + 3297280;  bf16t* Wk    = WBF + 3299328;
  bf16t* Wv    = WBF + 3301376;

  auto cdiv = [](long a, long b) { return (unsigned)((a + b - 1) / b); };

  // 1) materialize windowed input + pooled input (bf16)
  k_build_cin0<<<cdiv(2L * 65536 * 129, 256), 256, 0, stream>>>(x, CIN0);
  k_build_cin1<<<cdiv(2L * 32769 * 65, 256), 256, 0, stream>>>(x, CIN1);

  // 2) repack weights
  auto prep = [&](const float* w, bf16t* dst, int co, int ci, int kk2, int kp) {
    k_prep_w<<<cdiv((long)co * kp, 256), 256, 0, stream>>>(w, dst, co, ci, kk2, kk2, kp);
  };
  prep(f0w[0], Wf0_1, 64, 1, 12, 160);   prep(f1w[0], Wf1_1, 64, 1, 12, 160);
  prep(f0w[1], Wf0_2, 128, 64, 8, 4096); prep(f1w[1], Wf1_2, 128, 64, 8, 4096);
  prep(f0w[2], Wf0_3, 256, 128, 5, 3200);prep(f1w[2], Wf1_3, 256, 128, 5, 3200);
  prep(f0w[3], Wf0_4, 128, 256, 3, 2304);prep(f1w[3], Wf1_4, 128, 256, 3, 2304);
  k_prep_w<<<cdiv(16 * 128, 256), 256, 0, stream>>>(qw, Wq, 16, 128, 1, 1, 128);
  k_prep_w<<<cdiv(16 * 128, 256), 256, 0, stream>>>(kw, Wk, 16, 128, 1, 1, 128);
  k_prep_w<<<cdiv(128 * 128, 256), 256, 0, stream>>>(vw, Wv, 128, 128, 1, 1, 128);

  // MT = number of 16-row Cout tiles per wave (B-fragment reuse factor)
#define CONV(CI, CO, KK, S, P, RELU, MT, IN, HI, WI, WPTR, BPTR, OUT, HO, WO)    \
  k_conv_wmma<CI, CO, KK, KK, S, P, RELU, MT>                                    \
      <<<dim3(cdiv((long)(HO) * (WO), 64), (CO) / (16 * (MT)), 2), 128, 0,       \
         stream>>>(IN, HI, WI, WPTR, BPTR, OUT, HO, WO)

  // 3) f0 conv stack
  CONV(1,   64, 12, 6, 6, true, 4, CIN0, 65536, 129, Wf0_1, f0b[0], L1_0, 10923, 22);
  CONV(64, 128,  8, 4, 4, true, 4, L1_0, 10923, 22,  Wf0_2, f0b[1], L2_0, 2731, 6);
  CONV(128,256,  5, 2, 2, true, 4, L2_0, 2731, 6,    Wf0_3, f0b[2], L3_0, 1366, 3);
  CONV(256,128,  3, 1, 1, true, 4, L3_0, 1366, 3,    Wf0_4, f0b[3], C0,   1366, 3);

  // 4) f1 conv stack (pooled path)
  CONV(1,   64, 12, 6, 6, true, 4, CIN1, 32769, 65,  Wf1_1, f1b[0], L1_1, 5462, 11);
  CONV(64, 128,  8, 4, 4, true, 4, L1_1, 5462, 11,   Wf1_2, f1b[1], L2_1, 1366, 3);
  CONV(128,256,  5, 2, 2, true, 4, L2_1, 1366, 3,    Wf1_3, f1b[2], L3_1, 683, 2);
  CONV(256,128,  3, 1, 1, true, 4, L3_1, 683, 2,     Wf1_4, f1b[3], C1,   683, 2);

  // 5) q/k/v 1x1 projections (WMMA GEMM over pixels)
  CONV(128, 16, 1, 1, 0, false, 1, C0, 1366, 3, Wq, qb, Q0, 1366, 3);
  CONV(128, 16, 1, 1, 0, false, 1, C0, 1366, 3, Wk, kb, K0, 1366, 3);
  CONV(128,128, 1, 1, 0, false, 4, C0, 1366, 3, Wv, vb, V0, 1366, 3);
  CONV(128, 16, 1, 1, 0, false, 1, C1, 683, 2,  Wq, qb, Q1, 683, 2);
  CONV(128, 16, 1, 1, 0, false, 1, C1, 683, 2,  Wk, kb, K1, 683, 2);
  CONV(128,128, 1, 1, 0, false, 4, C1, 683, 2,  Wv, vb, V1, 683, 2);
#undef CONV

  // 6) CCA + channel means (featacc must be re-zeroed every call)
  k_zero<<<1, 512, 0, stream>>>(FEAT, 512);
  {
    int H = 1366, W = 3;
    size_t smem = (size_t)(16 + H + W + 256) * sizeof(float);
    k_cca<128, 16><<<dim3(H * W, 2), 256, smem, stream>>>(
        C0, Q0, K0, V0, gm, FEAT, H, W, 0);
  }
  {
    int H = 683, W = 2;
    size_t smem = (size_t)(16 + H + W + 256) * sizeof(float);
    k_cca<128, 16><<<dim3(H * W, 2), 256, smem, stream>>>(
        C1, Q1, K1, V1, gm, FEAT, H, W, 128);
  }

  // 7) classifier -> (2,2)
  k_classify<<<1, 32, 0, stream>>>(FEAT, cw, cb, (float*)d_out, 1366 * 3, 683 * 2);
}
